// UpdateNodeGlobal_24927990186022
// MI455X (gfx1250) — compile-verified, matrix operand order checked by reading, not run on hardware
//
#include <hip/hip_runtime.h>
#include <cstddef>

// ---------------------------------------------------------------------------
// Equivariant GNN layer for MI455X (gfx1250, wave32).
// All matmuls on v_wmma_f32_16x16x32_bf16 (f32 accumulate): bf16 WMMA puts the
// ~24 GFLOP of this layer at the 23.3 TB/s HBM roofline (~10us), where fp32
// WMMA (16x16x4) would be ~7x compute-bound. Weights are pre-packed once per
// launch into d_ws as bf16 N-major so every WMMA B-fragment is two b128 loads;
// A-fragments load as b128 runs. Per-tile Wigner blocks are fetched with the
// Tensor Data Mover (tensor_load_to_lds, waited with s_wait_tensorcnt) and
// overlap the K=128 WMMA phase. Edge messages scatter with
// global_atomic_add_f32 into agg[N,240] (d_ws), avoiding a 300MB msg buffer.
// ---------------------------------------------------------------------------

#define NN   10000
#define EE   160000
#define DIMF 240
#define LATD 128
#define CPLD 128
#define NTY  95

typedef __attribute__((ext_vector_type(16))) __bf16 v16bf;
typedef __attribute__((ext_vector_type(8)))  __bf16 v8bf;
typedef __attribute__((ext_vector_type(8)))  float  v8f;
typedef __attribute__((ext_vector_type(4)))  float  f32x4;
typedef __attribute__((ext_vector_type(4)))  unsigned int u32x4;
typedef __attribute__((ext_vector_type(8)))  int    i32x8;
typedef __attribute__((ext_vector_type(4)))  int    i32x4;

// ---- packed-weight element offsets in d_ws (after agg), bf16, N x Kp ------
enum {
  PW_Ws = 0,        // 160 x 128
  PW_W0 = 20480,    // 112 x 128
  PW_Wenv = 34816,  // 112 x 128
  PW_W1 = 49152,    // 32 x 64
  PW_W2 = 51200,    // 16 x 32
  PW_P0 = 51712,    // 64 x 64
  PW_P1 = 55808,    // 32 x 32
  PW_P2 = 56832,    // 16 x 32 (K padded 16->32 with zeros)
  PW_cW1 = 57344,   // 128 x 128
  PW_cW2 = 73728,   // 112 x 128
  PW_gsW = 88064,   // 128 x 64
  PW_mW1 = 96256,   // 128 x 256
  PW_mW2 = 129024,  // 128 x 128
  PW_rW0 = 145408,  // 64 x 64
  PW_rW1 = 149504,  // 32 x 32
  PW_rW2 = 150528,  // 16 x 32 (padded)
  PW_ohw0 = 151040, // 64 x 96 (K padded 95->96)
  PW_ohw1 = 157184, // 32 x 96
  PW_ohw2 = 160256, // 16 x 96
  PW_TOTAL = 161792
};

// ---- fragment helpers -----------------------------------------------------
// 16-bit A-frag (16x32) per ISA 7.12.2: per lane, elements 0..7 cover
// K = k0 + 8*half + {0..7}; elements 8..15 cover K = k0 + 16 + 8*half + {0..7}.
__device__ __forceinline__ int koff_map(int v, int h) {
  return ((v & 8) << 1) + (v & 7) + (h << 3);
}

__device__ __forceinline__ v16bf cat8(v8bf lo, v8bf hi) {
  return __builtin_shufflevector(lo, hi, 0, 1, 2, 3, 4, 5, 6, 7, 8, 9, 10, 11,
                                 12, 13, 14, 15);
}

__device__ __forceinline__ v8bf cvt8(const f32x4* p) {
  f32x4 a = p[0], b = p[1];
  v8bf r;
  r[0] = (__bf16)a[0]; r[1] = (__bf16)a[1]; r[2] = (__bf16)a[2]; r[3] = (__bf16)a[3];
  r[4] = (__bf16)b[0]; r[5] = (__bf16)b[1]; r[6] = (__bf16)b[2]; r[7] = (__bf16)b[3];
  return r;
}

// A-frag from a 16B-aligned f32 row in global memory (4x b128 + cvt)
__device__ __forceinline__ v16bf frag_a_row_f32(const float* row, int k0, int hf) {
  return cat8(cvt8((const f32x4*)(row + k0 + hf * 8)),
              cvt8((const f32x4*)(row + k0 + 16 + hf * 8)));
}
// A-frag from bf16 LDS row (2x ds_load_b128)
__device__ __forceinline__ v16bf frag_a_lds_bf16(const __bf16* base, int hf) {
  return cat8(*(const v8bf*)(base + hf * 8), *(const v8bf*)(base + 16 + hf * 8));
}
// A-frag from f32 LDS row (4x ds_load_b128 + cvt)
__device__ __forceinline__ v16bf frag_a_lds_f32(const float* base, int hf) {
  return cat8(cvt8((const f32x4*)(base + hf * 8)),
              cvt8((const f32x4*)(base + 16 + hf * 8)));
}
// B-frag from packed bf16 weights (N x Kp, N-major): 2x global b128 per lane
__device__ __forceinline__ v16bf frag_b(const __bf16* Bt, int Kp, int n0,
                                        int k0, int lane) {
  const __bf16* p = Bt + (size_t)(n0 + (lane & 15)) * Kp + k0 + (lane >> 4) * 8;
  return cat8(*(const v8bf*)p, *(const v8bf*)(p + 16));
}
// generic elementwise A-frag (strided / unaligned sources only)
template <typename F>
__device__ __forceinline__ v16bf frag_a_gen(int lane, F f) {
  const int m = lane & 15, h = lane >> 4;
  v16bf a;
#pragma unroll
  for (int v = 0; v < 16; ++v) a[v] = (__bf16)f(m, koff_map(v, h));
  return a;
}

__device__ __forceinline__ v8f wmma_bf16(v16bf a, v16bf b, v8f c) {
  return __builtin_amdgcn_wmma_f32_16x16x32_bf16(false, a, false, b, (short)0, c,
                                                 false, false);
}

__device__ __forceinline__ float silu_f(float x) { return x / (1.f + __expf(-x)); }
__device__ __forceinline__ float sigm_f(float x) { return 1.f / (1.f + __expf(-x)); }
// in-wave LDS RAW fence (wave32 lockstep; LDS ops are in-order per wave)
__device__ __forceinline__ void lds_fence() {
  asm volatile("s_wait_dscnt 0x0" ::: "memory");
}

// ---- Tensor Data Mover: async 1-D global->LDS copy of `nelem` dwords ------
// D# per CDNA5 ISA: group0 = {count=1, lds_addr,
// global_addr, type=2}; group1 = {data_size=4B, tensor_dim0=L, tensor_dim1=1,
// tile_dim0=L, tile_dim1=1, stride0=L}. Descriptors are wave-uniform
// (readfirstlane -> SGPR groups). Tracked by TENSORcnt. This toolchain's
// builtin takes the 6-arg form (g0, g1, g2, g3, extra group, cpol).
__device__ __forceinline__ void tdm_load_1d(unsigned lds_addr, const void* gptr,
                                            unsigned nelem) {
  unsigned long long ga = (unsigned long long)(size_t)gptr;
  u32x4 g0;
  g0[0] = (unsigned)__builtin_amdgcn_readfirstlane(1u);
  g0[1] = (unsigned)__builtin_amdgcn_readfirstlane(lds_addr);
  g0[2] = (unsigned)__builtin_amdgcn_readfirstlane((unsigned)ga);
  g0[3] = (unsigned)__builtin_amdgcn_readfirstlane(
      ((unsigned)(ga >> 32) & 0x01FFFFFFu) | 0x80000000u);  // type=2 ("image")
  i32x8 g1;
  g1[0] = __builtin_amdgcn_readfirstlane(0x20000);  // data_size code 2 = 4B
  g1[1] = __builtin_amdgcn_readfirstlane((int)((nelem & 0xffffu) << 16));
  g1[2] = __builtin_amdgcn_readfirstlane((int)((nelem >> 16) | (1u << 16)));
  g1[3] = __builtin_amdgcn_readfirstlane((int)((nelem & 0xffffu) << 16));
  g1[4] = __builtin_amdgcn_readfirstlane(1);        // tile_dim1=1, tile_dim2=0
  g1[5] = __builtin_amdgcn_readfirstlane((int)nelem);  // tensor_dim0_stride
  g1[6] = 0;
  g1[7] = 0;
  i32x4 z4 = {0, 0, 0, 0};
  i32x8 z8 = {0, 0, 0, 0, 0, 0, 0, 0};
  __builtin_amdgcn_tensor_load_to_lds(g0, g1, z4, z4, z8, 0);
}
__device__ __forceinline__ void tdm_wait() {
  __builtin_amdgcn_s_wait_tensorcnt((short)0);
  asm volatile("" ::: "memory");
}

// ======================= weight pack kernel ================================
__global__ __launch_bounds__(256) void pack_weights_kernel(
    const float* __restrict__ Ws, const float* __restrict__ W0,
    const float* __restrict__ Wenv, const float* __restrict__ W1,
    const float* __restrict__ W2, const float* __restrict__ P0,
    const float* __restrict__ P1, const float* __restrict__ P2,
    const float* __restrict__ cW1, const float* __restrict__ cW2,
    const float* __restrict__ gsW, const float* __restrict__ mW1,
    const float* __restrict__ mW2, const float* __restrict__ rW0,
    const float* __restrict__ rW1, const float* __restrict__ rW2,
    const float* __restrict__ ohw0, const float* __restrict__ ohw1,
    const float* __restrict__ ohw2, __bf16* __restrict__ dst) {
  const float* srcs[19] = {Ws, W0, Wenv, W1, W2, P0, P1, P2, cW1, cW2,
                           gsW, mW1, mW2, rW0, rW1, rW2, ohw0, ohw1, ohw2};
  const int KS[19] = {128, 128, 128, 64, 32, 64, 32, 16, 128, 128,
                      64, 256, 128, 64, 32, 16, 95, 95, 95};
  const int NS[19] = {160, 112, 112, 32, 16, 64, 32, 16, 128, 112,
                      128, 128, 128, 64, 32, 16, 64, 32, 16};
  const int KP[19] = {128, 128, 128, 64, 32, 64, 32, 32, 128, 128,
                      64, 256, 128, 64, 32, 32, 96, 96, 96};
  const int OFF[20] = {0,      20480,  34816,  49152,  51200,  51712,  55808,
                       56832,  57344,  73728,  88064,  96256,  129024, 145408,
                       149504, 150528, 151040, 157184, 160256, 161792};
  for (int idx = blockIdx.x * 256 + threadIdx.x; idx < PW_TOTAL;
       idx += gridDim.x * 256) {
    int mi = 0;
    while (idx >= OFF[mi + 1]) ++mi;
    int rel = idx - OFF[mi];
    int kp = KP[mi];
    int n = rel / kp, k = rel % kp;
    float v = 0.f;
    if (k < KS[mi])
      v = (mi >= 16) ? srcs[mi][n * KS[mi] + k]   // ohw*: stored N x K
                     : srcs[mi][k * NS[mi] + n];  // others: stored K x N
    dst[idx] = (__bf16)v;
  }
}

// ======================= edge kernel =======================================
struct alignas(16) EdgeWaveLds {
  float  D1c[16 * 9];
  float  D2c[16 * 25];
  int    src[16];
  __bf16 vstage[3 * 16 * 64];  // reused: v1_r / v1_g / v2_r / v2_g(pad32)
  __bf16 sact[16 * 64];
};

__global__ __launch_bounds__(128) void edge_msg_kernel(
    const float* __restrict__ nodef, const float* __restrict__ edgef,
    const float* __restrict__ lat, const float* __restrict__ D1,
    const float* __restrict__ D2, const float* __restrict__ Pb0,
    const int* __restrict__ srcidx, const __bf16* __restrict__ WP,
    float* __restrict__ agg) {
  __shared__ EdgeWaveLds sw[4];
  const int wave = threadIdx.x >> 5;
  const int lane = threadIdx.x & 31;
  EdgeWaveLds& S = sw[wave];
  const int tile = blockIdx.x * 4 + wave;
  const int e0 = tile * 16;
  const int hf = lane >> 4;
  const int l15 = lane & 15;

  // TDM async copy of this tile's Wigner blocks (contiguous f32 runs) into
  // LDS; completes behind the K=128 WMMA phase below. Static shared block
  // starts at LDS offset 0, so the per-wave LDS address is structural.
  const unsigned ldsw = (unsigned)(wave * sizeof(EdgeWaveLds));
  tdm_load_1d(ldsw + 0u, (const void*)(D1 + (size_t)e0 * 9), 16u * 9u);
  tdm_load_1d(ldsw + 576u, (const void*)(D2 + (size_t)e0 * 25), 16u * 25u);

  if (lane == 0) {  // pull the next tile toward the WGP while we compute
    __builtin_prefetch(lat + (size_t)(e0 + 64) * LATD, 0, 0);
    __builtin_prefetch(edgef + (size_t)(e0 + 64) * DIMF, 0, 0);
  }
  if (lane < 16) S.src[lane] = srcidx[e0 + lane];
  lds_fence();

  // A fragments: latents (16x128) and s_in = [ns[src] | es] (16x128)
  const float* rowL = lat + (size_t)(e0 + l15) * LATD;
  const float* rowN = nodef + (size_t)S.src[l15] * DIMF;
  const float* rowE = edgef + (size_t)(e0 + l15) * DIMF;
  v16bf Alat[4], Asin[4];
#pragma unroll
  for (int ks = 0; ks < 4; ++ks) Alat[ks] = frag_a_row_f32(rowL, ks * 32, hf);
  Asin[0] = frag_a_row_f32(rowN, 0, hf);
  Asin[1] = frag_a_row_f32(rowN, 32, hf);
  Asin[2] = frag_a_row_f32(rowE, 0, hf);
  Asin[3] = frag_a_row_f32(rowE, 32, hf);

  v8f wreg[7], gate[3], scv[3];
#pragma unroll
  for (int t = 0; t < 7; ++t) {  // 112 cols: sc, s_m, env in lock-step tiles
    v8f sc = {}, sm = {}, wv = {};
#pragma unroll
    for (int ks = 0; ks < 4; ++ks) {
      sc = wmma_bf16(Alat[ks], frag_b(WP + PW_Ws, 128, t * 16, ks * 32, lane), sc);
      sm = wmma_bf16(Asin[ks], frag_b(WP + PW_W0, 128, t * 16, ks * 32, lane), sm);
      wv = wmma_bf16(Alat[ks], frag_b(WP + PW_Wenv, 128, t * 16, ks * 32, lane), wv);
    }
    wreg[t] = wv;
#pragma unroll
    for (int r = 0; r < 8; ++r) sm[r] *= sc[r];
    if (t < 4) {  // s_act = silu(s_m[:, :64]) -> staged bf16 for post GEMM
#pragma unroll
      for (int r = 0; r < 8; ++r)
        S.sact[(hf * 8 + r) * 64 + t * 16 + l15] = (__bf16)silu_f(sm[r]);
    } else {      // gates = sigmoid(s_m[:, 64:112]) kept in D-frag registers
#pragma unroll
      for (int r = 0; r < 8; ++r) sm[r] = sigm_f(sm[r]);
      gate[t - 4] = sm;
    }
  }
#pragma unroll
  for (int t = 0; t < 3; ++t) {  // sc[:, 112:160] (v1/v2 channel scales)
    v8f sc = {};
#pragma unroll
    for (int ks = 0; ks < 4; ++ks)
      sc = wmma_bf16(Alat[ks],
                     frag_b(WP + PW_Ws, 128, 112 + t * 16, ks * 32, lane), sc);
    scv[t] = sc;
  }
  tdm_wait();  // Wigner blocks now resident in LDS
  lds_fence();

  // ---- L=1 path: rotate (D1), GEMM 64->32, scale, rotate back, gate ----
  for (int idx = lane; idx < 16 * 64; idx += 32) {
    int m = idx >> 6, c = idx & 63;
    const float* p = (c < 32) ? &nodef[S.src[m] * DIMF + 64 + c * 3]
                              : &edgef[(e0 + m) * DIMF + 64 + (c - 32) * 3];
    float x0 = p[0], x1 = p[1], x2 = p[2];
    const float* Dm = &S.D1c[m * 9];
#pragma unroll
    for (int i = 0; i < 3; ++i)
      S.vstage[i * 1024 + idx] =
          (__bf16)(Dm[i * 3 + 0] * x0 + Dm[i * 3 + 1] * x1 + Dm[i * 3 + 2] * x2);
  }
  lds_fence();

  v8f v1acc[3][2];
#pragma unroll
  for (int i = 0; i < 3; ++i)
#pragma unroll
    for (int t = 0; t < 2; ++t) {
      v8f acc = {};
#pragma unroll
      for (int ks = 0; ks < 2; ++ks)
        acc = wmma_bf16(
            frag_a_lds_bf16(&S.vstage[i * 1024 + l15 * 64 + ks * 32], hf),
            frag_b(WP + PW_W1, 64, t * 16, ks * 32, lane), acc);
#pragma unroll
      for (int r = 0; r < 8; ++r) acc[r] *= scv[t][r];
      v1acc[i][t] = acc;
    }
#pragma unroll
  for (int t = 0; t < 2; ++t) {  // rotate back + gate -> stage v1_g
    int o = t * 16 + l15;
#pragma unroll
    for (int r = 0; r < 8; ++r) {
      int m = hf * 8 + r;
      const float* Dm = &S.D1c[m * 9];
      float g = gate[t][r];
      float x0 = v1acc[0][t][r], x1 = v1acc[1][t][r], x2 = v1acc[2][t][r];
#pragma unroll
      for (int j = 0; j < 3; ++j) {
        float vj = Dm[j * 3 + 0] * x0 + Dm[j * 3 + 1] * x1 + Dm[j * 3 + 2] * x2;
        S.vstage[j * 512 + m * 32 + o] = (__bf16)(vj * g);
      }
    }
  }
  lds_fence();
#pragma unroll
  for (int j = 0; j < 3; ++j)
#pragma unroll
    for (int t = 0; t < 2; ++t) {  // post_W1, env scale, scatter
      v8f acc = {};
      acc = wmma_bf16(frag_a_lds_bf16(&S.vstage[j * 512 + l15 * 32], hf),
                      frag_b(WP + PW_P1, 32, t * 16, 0, lane), acc);
      int o = t * 16 + l15;
#pragma unroll
      for (int r = 0; r < 8; ++r) {
        int m = hf * 8 + r;
        atomicAdd(&agg[S.src[m] * DIMF + 64 + o * 3 + j],
                  acc[r] * wreg[4 + t][r] * 0.25f);
      }
    }
  lds_fence();

  // ---- L=2 path: rotate (D2), GEMM 32->16, scale, rotate back, gate ----
  for (int idx = lane; idx < 16 * 32; idx += 32) {
    int m = idx >> 5, c = idx & 31;
    const float* p = (c < 16) ? &nodef[S.src[m] * DIMF + 160 + c * 5]
                              : &edgef[(e0 + m) * DIMF + 160 + (c - 16) * 5];
    float x[5];
#pragma unroll
    for (int j = 0; j < 5; ++j) x[j] = p[j];
    const float* Dm = &S.D2c[m * 25];
#pragma unroll
    for (int i = 0; i < 5; ++i) {
      float a = 0.f;
#pragma unroll
      for (int j = 0; j < 5; ++j) a += Dm[i * 5 + j] * x[j];
      S.vstage[i * 512 + idx] = (__bf16)a;
    }
  }
  lds_fence();
  v8f v2acc[5];
#pragma unroll
  for (int i = 0; i < 5; ++i) {
    v8f acc = {};
    acc = wmma_bf16(frag_a_lds_bf16(&S.vstage[i * 512 + l15 * 32], hf),
                    frag_b(WP + PW_W2, 32, 0, 0, lane), acc);
#pragma unroll
    for (int r = 0; r < 8; ++r) acc[r] *= scv[2][r];
    v2acc[i] = acc;
  }
  {
    int o = l15;
#pragma unroll
    for (int r = 0; r < 8; ++r) {
      int m = hf * 8 + r;
      const float* Dm = &S.D2c[m * 25];
      float g = gate[2][r];
      float x[5];
#pragma unroll
      for (int i = 0; i < 5; ++i) x[i] = v2acc[i][r];
#pragma unroll
      for (int j = 0; j < 5; ++j) {
        float vj = 0.f;
#pragma unroll
        for (int i = 0; i < 5; ++i) vj += Dm[j * 5 + i] * x[i];
        S.vstage[j * 512 + m * 32 + o] = (__bf16)(vj * g);
        S.vstage[j * 512 + m * 32 + 16 + o] = (__bf16)0.f;  // K pad 16->32
      }
    }
  }
  lds_fence();
#pragma unroll
  for (int j = 0; j < 5; ++j) {  // post_W2 (packed weights are K-padded)
    v8f acc = {};
    acc = wmma_bf16(frag_a_lds_bf16(&S.vstage[j * 512 + l15 * 32], hf),
                    frag_b(WP + PW_P2, 32, 0, 0, lane), acc);
    int o = l15;
#pragma unroll
    for (int r = 0; r < 8; ++r) {
      int m = hf * 8 + r;
      atomicAdd(&agg[S.src[m] * DIMF + 160 + o * 5 + j],
                acc[r] * wreg[6][r] * 0.25f);
    }
  }

  // ---- scalar post: s_act @ post_W0 + b0, env scale, scatter ----
#pragma unroll
  for (int t = 0; t < 4; ++t) {
    v8f acc = {};
#pragma unroll
    for (int ks = 0; ks < 2; ++ks)
      acc = wmma_bf16(frag_a_lds_bf16(&S.sact[l15 * 64 + ks * 32], hf),
                      frag_b(WP + PW_P0, 64, t * 16, ks * 32, lane), acc);
    int n = t * 16 + l15;
#pragma unroll
    for (int r = 0; r < 8; ++r) {
      int m = hf * 8 + r;
      atomicAdd(&agg[S.src[m] * DIMF + n],
                (acc[r] + Pb0[n]) * wreg[t][r] * 0.25f);
    }
  }
}

// ======================= node kernel =======================================
struct alignas(16) NodeWaveLds {
  float nf[16 * 240];
  union {
    __bf16 hst[16 * 256];  // bf16 activation staging (h1 / h / silu(h@W1))
    float  mcf[16 * 128];  // f32 mc for LayerNorm (reused after hst dies)
  };
  float cg[16 * 112];      // coeffs, later one-hot gains
  float stats[32];
};

__global__ __launch_bounds__(64) void node_update_kernel(
    const float* __restrict__ agg, const float* __restrict__ mcpl,
    const float* __restrict__ cb1, const float* __restrict__ cb2,
    const float* __restrict__ gsb, const float* __restrict__ mb1,
    const float* __restrict__ mb2, const float* __restrict__ lng,
    const float* __restrict__ lnb, const float* __restrict__ nodef,
    const float* __restrict__ rb0, const float* __restrict__ onehot,
    const __bf16* __restrict__ WP, float* __restrict__ out) {
  __shared__ NodeWaveLds sw[2];
  const int wave = threadIdx.x >> 5;
  const int lane = threadIdx.x & 31;
  const int tile = blockIdx.x * 2 + wave;
  const int n0 = tile * 16;
  if (n0 >= NN) return;  // no cross-wave barriers anywhere in this kernel
  NodeWaveLds& S = sw[wave];
  const int hf = lane >> 4;
  const int l15 = lane & 15;

  // ---- coeffs = silu(mc@cW1+b1)@cW2+b2 ----
  const float* rowM = mcpl + (size_t)(n0 + l15) * CPLD;
  v16bf Amc[4];
#pragma unroll
  for (int ks = 0; ks < 4; ++ks) Amc[ks] = frag_a_row_f32(rowM, ks * 32, hf);
#pragma unroll
  for (int t = 0; t < 8; ++t) {
    v8f acc = {};
#pragma unroll
    for (int ks = 0; ks < 4; ++ks)
      acc = wmma_bf16(Amc[ks], frag_b(WP + PW_cW1, 128, t * 16, ks * 32, lane),
                      acc);
    int n = t * 16 + l15;
#pragma unroll
    for (int r = 0; r < 8; ++r)
      S.hst[(hf * 8 + r) * 256 + n] = (__bf16)silu_f(acc[r] + cb1[n]);
  }
  lds_fence();
#pragma unroll
  for (int t = 0; t < 7; ++t) {
    v8f acc = {};
#pragma unroll
    for (int ks = 0; ks < 4; ++ks)
      acc = wmma_bf16(frag_a_lds_bf16(&S.hst[l15 * 256 + ks * 32], hf),
                      frag_b(WP + PW_cW2, 128, t * 16, ks * 32, lane), acc);
    int n = t * 16 + l15;
#pragma unroll
    for (int r = 0; r < 8; ++r)
      S.cg[(hf * 8 + r) * 112 + n] = acc[r] + cb2[n];
  }
  lds_fence();

  // ---- new_nf = agg * coeffs[:, SCALE_IDX] ----
  for (int idx = lane; idx < 16 * 240; idx += 32) {
    int m = idx / 240, d = idx % 240;
    int ch = (d < 64) ? d : (d < 160 ? 64 + (d - 64) / 3 : 96 + (d - 160) / 5);
    S.nf[idx] = agg[(n0 + m) * DIMF + d] * S.cg[m * 112 + ch];
  }
  lds_fence();

  // ---- scal = nf[:, :64]@gsW + gsb ; h = [scal | msg_cpl] ----
#pragma unroll
  for (int t = 0; t < 8; ++t) {
    v8f acc = {};
#pragma unroll
    for (int ks = 0; ks < 2; ++ks)
      acc = wmma_bf16(frag_a_lds_f32(&S.nf[l15 * 240 + ks * 32], hf),
                      frag_b(WP + PW_gsW, 64, t * 16, ks * 32, lane), acc);
    int n = t * 16 + l15;
#pragma unroll
    for (int r = 0; r < 8; ++r)
      S.hst[(hf * 8 + r) * 256 + n] = (__bf16)(acc[r] + gsb[n]);
  }
  for (int idx = lane; idx < 16 * 128; idx += 32) {
    int m = idx >> 7, c = idx & 127;
    S.hst[m * 256 + 128 + c] = (__bf16)mcpl[(n0 + m) * 128 + c];
  }
  lds_fence();

  // ---- upd = silu(h@mW1+mb1)@mW2+mb2 ; mc = msg_cpl + upd ----
  v8f u[8];
#pragma unroll
  for (int t = 0; t < 8; ++t) {
    v8f acc = {};
#pragma unroll
    for (int ks = 0; ks < 8; ++ks)
      acc = wmma_bf16(frag_a_lds_bf16(&S.hst[l15 * 256 + ks * 32], hf),
                      frag_b(WP + PW_mW1, 256, t * 16, ks * 32, lane), acc);
    int n = t * 16 + l15;
#pragma unroll
    for (int r = 0; r < 8; ++r) u[t][r] = silu_f(acc[r] + mb1[n]);
  }
#pragma unroll
  for (int t = 0; t < 8; ++t) {
    int n = t * 16 + l15;
#pragma unroll
    for (int r = 0; r < 8; ++r) S.hst[(hf * 8 + r) * 256 + n] = (__bf16)u[t][r];
  }
  lds_fence();
  v8f u2[8];
#pragma unroll
  for (int t = 0; t < 8; ++t) {
    v8f acc = {};
#pragma unroll
    for (int ks = 0; ks < 4; ++ks)
      acc = wmma_bf16(frag_a_lds_bf16(&S.hst[l15 * 256 + ks * 32], hf),
                      frag_b(WP + PW_mW2, 128, t * 16, ks * 32, lane), acc);
    int n = t * 16 + l15;
#pragma unroll
    for (int r = 0; r < 8; ++r)
      u2[t][r] = acc[r] + mb2[n] + mcpl[(n0 + hf * 8 + r) * 128 + n];
  }
  lds_fence();
#pragma unroll
  for (int t = 0; t < 8; ++t) {  // hst is dead; reuse bytes as f32 mc
    int n = t * 16 + l15;
#pragma unroll
    for (int r = 0; r < 8; ++r) S.mcf[(hf * 8 + r) * 128 + n] = u2[t][r];
  }
  lds_fence();
  if (lane < 16) {  // per-row LayerNorm stats
    float s = 0.f, s2 = 0.f;
    for (int c = 0; c < 128; ++c) {
      float v = S.mcf[lane * 128 + c];
      s += v; s2 += v * v;
    }
    float mu = s * (1.f / 128.f);
    float var = s2 * (1.f / 128.f) - mu * mu;
    S.stats[lane * 2] = mu;
    S.stats[lane * 2 + 1] = rsqrtf(var + 1e-5f);
  }
  lds_fence();
  for (int idx = lane; idx < 16 * 128; idx += 32) {
    int m = idx >> 7, c = idx & 127;
    float v = (S.mcf[idx] - S.stats[m * 2]) * S.stats[m * 2 + 1] * lng[c] + lnb[c];
    out[(size_t)NN * DIMF + (n0 + m) * 128 + c] = v;
  }

  // ---- one-hot gains g0/g1/g2 (packed weights K-padded 95->96) ----
  v16bf Aoh[3];
#pragma unroll
  for (int ks = 0; ks < 3; ++ks)
    Aoh[ks] = frag_a_gen(lane, [&](int m, int ko) {  // 95-f32 rows: unaligned
      int k = ks * 32 + ko;
      return (k < NTY) ? onehot[(n0 + m) * NTY + k] : 0.f;
    });
  const float inv = 0.10259783520851541f;  // 1/sqrt(95)
#pragma unroll
  for (int t = 0; t < 4; ++t) {
    v8f acc = {};
#pragma unroll
    for (int ks = 0; ks < 3; ++ks)
      acc = wmma_bf16(Aoh[ks], frag_b(WP + PW_ohw0, 96, t * 16, ks * 32, lane),
                      acc);
    int n = t * 16 + l15;
#pragma unroll
    for (int r = 0; r < 8; ++r) S.cg[(hf * 8 + r) * 112 + n] = acc[r] * inv;
  }
#pragma unroll
  for (int t = 0; t < 2; ++t) {
    v8f acc = {};
#pragma unroll
    for (int ks = 0; ks < 3; ++ks)
      acc = wmma_bf16(Aoh[ks], frag_b(WP + PW_ohw1, 96, t * 16, ks * 32, lane),
                      acc);
    int n = t * 16 + l15;
#pragma unroll
    for (int r = 0; r < 8; ++r) S.cg[(hf * 8 + r) * 112 + 64 + n] = acc[r] * inv;
  }
  {
    v8f acc = {};
#pragma unroll
    for (int ks = 0; ks < 3; ++ks)
      acc = wmma_bf16(Aoh[ks], frag_b(WP + PW_ohw2, 96, 0, ks * 32, lane), acc);
#pragma unroll
    for (int r = 0; r < 8; ++r) S.cg[(hf * 8 + r) * 112 + 96 + l15] = acc[r] * inv;
  }
  lds_fence();

  // ---- residual GEMMs, combine, fused one-hot gain, final store ----
  const float c_new = 0.4472135954999579f;  // sigmoid(0)*rsqrt(1.25)
  const float c_old = 0.8944271909999159f;  // rsqrt(1.25)
  const float* rowNF = nodef + (size_t)(n0 + l15) * DIMF;
  v16bf Ans[2];
  Ans[0] = frag_a_row_f32(rowNF, 0, hf);
  Ans[1] = frag_a_row_f32(rowNF, 32, hf);
#pragma unroll
  for (int t = 0; t < 4; ++t) {
    v8f acc = {};
#pragma unroll
    for (int ks = 0; ks < 2; ++ks)
      acc = wmma_bf16(Ans[ks], frag_b(WP + PW_rW0, 64, t * 16, ks * 32, lane),
                      acc);
    int n = t * 16 + l15;
#pragma unroll
    for (int r = 0; r < 8; ++r) {
      int m = hf * 8 + r;
      float res = acc[r] + rb0[n];
      float v = c_new * S.nf[m * 240 + n] + c_old * res;
      out[(n0 + m) * DIMF + n] = v * (1.f + S.cg[m * 112 + n]);
    }
  }
#pragma unroll
  for (int i = 0; i < 3; ++i) {
    v16bf A = frag_a_gen(lane, [&](int m, int ko) {  // stride-3 gather
      return nodef[(n0 + m) * DIMF + 64 + ko * 3 + i];
    });
#pragma unroll
    for (int t = 0; t < 2; ++t) {
      v8f acc = {};
      acc = wmma_bf16(A, frag_b(WP + PW_rW1, 32, t * 16, 0, lane), acc);
      int o = t * 16 + l15;
#pragma unroll
      for (int r = 0; r < 8; ++r) {
        int m = hf * 8 + r;
        int d = 64 + o * 3 + i;
        float v = c_new * S.nf[m * 240 + d] + c_old * acc[r];
        out[(n0 + m) * DIMF + d] = v * (1.f + S.cg[m * 112 + 64 + o]);
      }
    }
  }
#pragma unroll
  for (int i = 0; i < 5; ++i) {
    v16bf A = frag_a_gen(lane, [&](int m, int ko) {  // stride-5 gather, K pad
      return (ko < 16) ? nodef[(n0 + m) * DIMF + 160 + ko * 5 + i] : 0.f;
    });
    v8f acc = {};
    acc = wmma_bf16(A, frag_b(WP + PW_rW2, 32, 0, 0, lane), acc);
    int o = l15;
#pragma unroll
    for (int r = 0; r < 8; ++r) {
      int m = hf * 8 + r;
      int d = 160 + o * 5 + i;
      float v = c_new * S.nf[m * 240 + d] + c_old * acc[r];
      out[(n0 + m) * DIMF + d] = v * (1.f + S.cg[m * 112 + 96 + o]);
    }
  }
}

// ======================= launch ============================================
extern "C" void kernel_launch(void* const* d_in, const int* in_sizes, int n_in,
                              void* d_out, int out_size, void* d_ws,
                              size_t ws_size, hipStream_t stream) {
  (void)in_sizes; (void)n_in; (void)out_size; (void)ws_size;
  const float* nodef  = (const float*)d_in[0];
  const float* edgef  = (const float*)d_in[1];
  const float* lat    = (const float*)d_in[2];
  const float* mcpl   = (const float*)d_in[3];
  const float* onehot = (const float*)d_in[4];
  const float* D1     = (const float*)d_in[5];
  const float* D2     = (const float*)d_in[6];
  const float* W0     = (const float*)d_in[7];
  const float* W1     = (const float*)d_in[8];
  const float* W2     = (const float*)d_in[9];
  const float* Ws     = (const float*)d_in[10];
  const float* Wenv   = (const float*)d_in[11];
  const float* P0     = (const float*)d_in[12];
  const float* Pb0    = (const float*)d_in[13];
  const float* P1     = (const float*)d_in[14];
  const float* P2     = (const float*)d_in[15];
  const float* rW0    = (const float*)d_in[16];
  const float* rb0    = (const float*)d_in[17];
  const float* rW1    = (const float*)d_in[18];
  const float* rW2    = (const float*)d_in[19];
  const float* cW1    = (const float*)d_in[20];
  const float* cb1    = (const float*)d_in[21];
  const float* cW2    = (const float*)d_in[22];
  const float* cb2    = (const float*)d_in[23];
  const float* gsW    = (const float*)d_in[24];
  const float* gsb    = (const float*)d_in[25];
  const float* mW1    = (const float*)d_in[26];
  const float* mb1    = (const float*)d_in[27];
  const float* mW2    = (const float*)d_in[28];
  const float* mb2    = (const float*)d_in[29];
  const float* lng    = (const float*)d_in[30];
  const float* lnb    = (const float*)d_in[31];
  const float* ohw0   = (const float*)d_in[32];
  const float* ohw1   = (const float*)d_in[33];
  const float* ohw2   = (const float*)d_in[34];
  const int*   eidx   = (const int*)d_in[35];  // row 0 = src
  float* out = (float*)d_out;
  float* agg = (float*)d_ws;                     // [N,240] f32 accumulator
  __bf16* wp = (__bf16*)(agg + (size_t)NN * DIMF);  // packed bf16 weights

  (void)hipMemsetAsync(agg, 0, (size_t)NN * DIMF * sizeof(float), stream);
  pack_weights_kernel<<<(PW_TOTAL + 255) / 256, 256, 0, stream>>>(
      Ws, W0, Wenv, W1, W2, P0, P1, P2, cW1, cW2, gsW, mW1, mW2, rW0, rW1, rW2,
      ohw0, ohw1, ohw2, wp);
  edge_msg_kernel<<<EE / 64, 128, 0, stream>>>(nodef, edgef, lat, D1, D2, Pb0,
                                               eidx, wp, agg);
  node_update_kernel<<<(NN / 16 + 1) / 2, 64, 0, stream>>>(
      agg, mcpl, cb1, cb2, gsb, mb1, mb2, lng, lnb, nodef, rb0, onehot, wp, out);
}
